// ud_MultiheadSoftmaxAttention_3118146257157
// MI455X (gfx1250) — matrix operand order repaired and checked by reading, not compiled, not run on hardware
//
#include <hip/hip_runtime.h>
#include <hip/hip_bf16.h>
#include <math.h>

// ---------------------------------------------------------------------------
// Problem constants
// ---------------------------------------------------------------------------
constexpr int Bsz = 2;
constexpr int T   = 2048;
constexpr int D   = 1024;
constexpr int H   = 16;
constexpr int HD  = 64;
constexpr int BT  = Bsz * T;   // 4096 tokens

typedef __attribute__((ext_vector_type(16))) _Float16 v16h;
typedef __attribute__((ext_vector_type(8)))  _Float16 v8h;
typedef __attribute__((ext_vector_type(4)))  _Float16 v4h;
typedef __attribute__((ext_vector_type(8)))  float    v8f;

__device__ __forceinline__ v16h cat8(v8h lo, v8h hi) {
    return __builtin_shufflevector(lo, hi, 0,1,2,3,4,5,6,7,8,9,10,11,12,13,14,15);
}

__device__ __forceinline__ v8f wmma_f16(v16h a, v16h b, v8f c) {
    // D = A(16x32 f16) * B(32x16 f16) + C(16x16 f32)
    return __builtin_amdgcn_wmma_f32_16x16x32_f16(
        /*neg_a=*/false, a, /*neg_b=*/false, b,
        /*c_mod=*/(short)0, c, /*reuse_a=*/false, /*reuse_b=*/false);
}

// DPP ROW_XMASK butterfly within each 16-lane row (VALU pipe, not LDS pipe).
template <int MASK>
__device__ __forceinline__ float dpp_xmask(float x) {
    int v = __builtin_amdgcn_update_dpp(0, __float_as_int(x),
                                        0x160 | MASK, 0xF, 0xF, true);
    return __int_as_float(v);
}
__device__ __forceinline__ float rowmax16(float x) {
    x = fmaxf(x, dpp_xmask<1>(x));
    x = fmaxf(x, dpp_xmask<2>(x));
    x = fmaxf(x, dpp_xmask<4>(x));
    x = fmaxf(x, dpp_xmask<8>(x));
    return x;
}

// CDNA5 async memory->LDS copy (no VGPR staging, tracked by ASYNCcnt).
__device__ __forceinline__ void async_ld_b128(unsigned lds_off, const void* gptr) {
    asm volatile("global_load_async_to_lds_b128 %0, %1, off"
                 :: "v"(lds_off), "v"(gptr) : "memory");
}
__device__ __forceinline__ void wait_async_le2() {
    asm volatile("s_wait_asynccnt 0x2" ::: "memory");
}

// ---------------------------------------------------------------------------
// f32 -> f16 conversion (vectorized x4)
// ---------------------------------------------------------------------------
__global__ __launch_bounds__(256) void cvt_f16x4(const float4* __restrict__ src,
                                                 v4h* __restrict__ dst, int n4) {
    int i = blockIdx.x * blockDim.x + threadIdx.x;
    if (i < n4) {
        float4 f = src[i];
        v4h o;
        o[0] = (_Float16)f.x; o[1] = (_Float16)f.y;
        o[2] = (_Float16)f.z; o[3] = (_Float16)f.w;
        dst[i] = o;
    }
}

// ---------------------------------------------------------------------------
// Per-token softmax bias terms:
//   shift = w_sigma*mean(logvar) + w_phi*(1 - mean(phi))
//   inv   = 1 / max(mean(phi), 1e-6)        (GAMMA == 1)
// ---------------------------------------------------------------------------
__global__ __launch_bounds__(256) void token_stats(const float* __restrict__ phi,
                                                   const float* __restrict__ logvar,
                                                   const float* __restrict__ wsig,
                                                   const float* __restrict__ wphi,
                                                   float* __restrict__ shiftv,
                                                   float* __restrict__ invv) {
    int i = blockIdx.x * blockDim.x + threadIdx.x;
    if (i >= BT) return;
    float sp = 0.f, sl = 0.f;
    #pragma unroll
    for (int f = 0; f < 8; ++f) { sp += phi[i*8+f]; sl += logvar[i*8+f]; }
    sp *= 0.125f; sl *= 0.125f;
    shiftv[i] = wsig[0]*sl + wphi[0]*(1.0f - sp);
    invv[i]   = 1.0f / fmaxf(sp, 1e-6f);
}

// ---------------------------------------------------------------------------
// GEMM: Y = X @ W^T + bias.  X: [4096,1024] f16, W: [1024(out),1024(in)] f16.
// Each wave: 32(M) x 64(N) tile, K-loop step 32, 8 v_wmma per step.
// MODE 0: write f16 [b,h,t,d] (Q/K)    MODE 1: write f16 [b,h,d,t] (V^T)
// MODE 2: write f32 [m,n] to d_out (output projection)
// ---------------------------------------------------------------------------
template <int MODE>
__global__ __launch_bounds__(128) void gemm_kernel(const _Float16* __restrict__ X,
                                                   const _Float16* __restrict__ W,
                                                   const float* __restrict__ bias,
                                                   _Float16* __restrict__ out16,
                                                   float* __restrict__ out32) {
    const int wid  = blockIdx.x * 4 + (threadIdx.x >> 5);
    const int tm   = wid >> 4;            // 128 tiles of 32 rows
    const int tn   = wid & 15;            // 16 tiles of 64 cols
    const int lane = threadIdx.x & 31;
    const int lm   = lane & 15;
    const int hl   = lane >> 4;

    v8f acc[2][4];
    #pragma unroll
    for (int i = 0; i < 2; ++i)
        #pragma unroll
        for (int j = 0; j < 4; ++j) acc[i][j] = (v8f){};

    const int rowA0 = tm * 32 + lm;
    const int colB0 = tn * 64 + lm;

    for (int k0 = 0; k0 < D; k0 += 32) {
        v16h a[2];
        #pragma unroll
        for (int i = 0; i < 2; ++i) {
            const _Float16* p = X + (size_t)(rowA0 + i*16) * D + k0 + hl*16;
            a[i] = cat8(*(const v8h*)p, *(const v8h*)(p + 8));
        }
        v16h b[4];
        #pragma unroll
        for (int j = 0; j < 4; ++j)
            b[j] = *(const v16h*)(W + (size_t)(colB0 + j*16) * D + k0 + hl*16);
        #pragma unroll
        for (int i = 0; i < 2; ++i)
            #pragma unroll
            for (int j = 0; j < 4; ++j)
                acc[i][j] = wmma_f16(a[i], b[j], acc[i][j]);
    }

    // Epilogue. C layout: VGPR r, lane L -> M = r + 8*(L>>4), N = L&15.
    #pragma unroll
    for (int j = 0; j < 4; ++j) {
        const int col = tn*64 + j*16 + lm;          // output feature n
        const float bcol = bias[col];
        const int hh = col >> 6, dd = col & 63;
        #pragma unroll
        for (int i = 0; i < 2; ++i)
            #pragma unroll
            for (int r = 0; r < 8; ++r) {
                const int row = tm*32 + i*16 + hl*8 + r;   // token index m
                const float y = acc[i][j][r] + bcol;
                if (MODE == 2) {
                    out32[(size_t)row * D + col] = y;
                } else {
                    const int bb = row >> 11, tt = row & (T - 1);
                    if (MODE == 0)
                        out16[(((size_t)(bb*H + hh))*T + tt)*HD + dd] = (_Float16)y;
                    else
                        out16[(((size_t)(bb*H + hh))*HD + dd)*T + tt] = (_Float16)y;
                }
            }
    }
}

// ---------------------------------------------------------------------------
// Flash-style attention. 4 waves per block share one (b,h); K/V tiles are
// double-buffered in LDS and filled with GLOBAL_LOAD_ASYNC_TO_LDS_B128 so
// tile kt+1 streams in while tile kt feeds the WMMAs. Online softmax in
// exp2 domain; row max via DPP ROW_XMASK (VALU); row sums via a 5th WMMA
// against an all-ones B fragment (matrix pipe).
// ---------------------------------------------------------------------------
__global__ __launch_bounds__(128) void attn_kernel(const _Float16* __restrict__ q16,
                                                   const _Float16* __restrict__ k16,
                                                   const _Float16* __restrict__ vT,
                                                   const float* __restrict__ shiftv,
                                                   const float* __restrict__ invv,
                                                   _Float16* __restrict__ attn16) {
    const int bh    = blockIdx.x >> 5;     // 32 blocks per (b,h): T/64 q-tiles
    const int qblk  = blockIdx.x & 31;
    const int wave  = threadIdx.x >> 5;
    const int lane  = threadIdx.x & 31;
    const int lm    = lane & 15;
    const int hl    = lane >> 4;
    const int tid   = threadIdx.x;
    const int bb    = bh >> 4;
    const int hh    = bh & 15;
    const int qbase = (qblk * 4 + wave) * 16;

    const _Float16* qp = q16 + (size_t)bh * T * HD;
    const _Float16* kp = k16 + (size_t)bh * T * HD;
    const _Float16* vp = vT  + (size_t)bh * HD * T;

    __shared__ __align__(32) _Float16 Kt[2][32 * 64];   // [key][d]   2x4 KB
    __shared__ __align__(32) _Float16 Vt[2][64 * 32];   // [d][key]   2x4 KB
    __shared__ __align__(32) _Float16 Pl[4][16 * 32];   // per-wave P 4 KB

    // Q as A fragment: two K=32 chunks of the 64-wide head dim.
    v16h aq[2];
    #pragma unroll
    for (int c = 0; c < 2; ++c) {
        const _Float16* p = qp + (size_t)(qbase + lm) * HD + c*32 + hl*16;
        aq[c] = cat8(*(const v8h*)p, *(const v8h*)(p + 8));
    }

    // All-ones B fragment: Lacc = P x ones accumulates row sums of P.
    v16h ones;
    #pragma unroll
    for (int i = 0; i < 16; ++i) ones[i] = (_Float16)1.0f;

    // Per-row constants in exp2 domain: s2' = S*c1 - c2  (log2e folded in)
    constexpr float LOG2E = 1.4426950408889634f;
    float c1[8], c2[8], mrow[8];
    #pragma unroll
    for (int r = 0; r < 8; ++r) {
        const int qrow = qbase + hl*8 + r;
        const float iv = invv[bb*T + qrow];
        const float sh = shiftv[bb*T + qrow];
        c1[r] = 0.125f * iv * LOG2E;       // 1/sqrt(64) and log2e folded in
        c2[r] = sh * iv * LOG2E;
        mrow[r] = -INFINITY;
    }
    v8f O[4], Lacc;
    #pragma unroll
    for (int c = 0; c < 4; ++c) O[c] = (v8f){};
    Lacc = (v8f){};

    // Cooperative-stage indices: one b128 per thread per tile for K and V.
    const int kKey = tid >> 2;            // 0..31
    const int kDc  = (tid & 3) * 16;      // 0,16,32,48
    const int vD   = tid >> 1;            // 0..63
    const int vKc  = (tid & 1) * 16;      // 0,16

    const unsigned kOff[2] = { (unsigned)(size_t)&Kt[0][kKey*64 + kDc],
                               (unsigned)(size_t)&Kt[1][kKey*64 + kDc] };
    const unsigned vOff[2] = { (unsigned)(size_t)&Vt[0][vD*32 + vKc],
                               (unsigned)(size_t)&Vt[1][vD*32 + vKc] };

    // Prologue: async-stage tile 0 into buffer 0.
    async_ld_b128(kOff[0], kp + (size_t)kKey*HD + kDc);
    async_ld_b128(vOff[0], vp + (size_t)vD*T + vKc);

    for (int kt = 0; kt < T/32; ++kt) {
        const int kb  = kt * 32;
        const int buf = kt & 1;
        const int nkb = (kb + 32) & (T - 1);   // wraps on last iter (benign)

        // Prefetch tile kt+1 into the other buffer (ASYNCcnt += 2).
        async_ld_b128(kOff[buf ^ 1], kp + (size_t)(nkb + kKey)*HD + kDc);
        async_ld_b128(vOff[buf ^ 1], vp + (size_t)vD*T + nkb + vKc);

        // Wait for tile kt (leave the 2 prefetch ops in flight), then sync.
        wait_async_le2();
        __syncthreads();

        const _Float16* Kc = Kt[buf];
        const _Float16* Vc = Vt[buf];

        // S = Q K^T for 16q x 32k (two 16x16 N-tiles, K over head dim)
        v8f S[2]; S[0] = (v8f){}; S[1] = (v8f){};
        #pragma unroll
        for (int s = 0; s < 2; ++s) {
            const int keyl = s*16 + lm;
            #pragma unroll
            for (int c = 0; c < 2; ++c) {
                v16h bk = *(const v16h*)(Kc + keyl*64 + c*32 + hl*16);
                S[s] = wmma_f16(aq[c], bk, S[s]);
            }
        }

        // Online softmax (exp2 domain). Row max via DPP butterflies.
        float p0[8], p1[8], alpha[8];
        #pragma unroll
        for (int r = 0; r < 8; ++r) {
            const float s0 = S[0][r] * c1[r] - c2[r];
            const float s1 = S[1][r] * c1[r] - c2[r];
            const float tmax = rowmax16(fmaxf(s0, s1));
            const float mn = fmaxf(mrow[r], tmax);
            alpha[r] = exp2f(mrow[r] - mn);
            p0[r] = exp2f(s0 - mn);
            p1[r] = exp2f(s1 - mn);
            mrow[r] = mn;
        }
        #pragma unroll
        for (int c = 0; c < 4; ++c)
            #pragma unroll
            for (int r = 0; r < 8; ++r)
                O[c][r] *= alpha[r];
        #pragma unroll
        for (int r = 0; r < 8; ++r)
            Lacc[r] *= alpha[r];

        // Stage P (16x32 f16, row-major) in this wave's LDS region.
        // Same-wave DS ops are in-order, so no barrier needed before reload.
        _Float16* Pw = Pl[wave];
        #pragma unroll
        for (int r = 0; r < 8; ++r) {
            Pw[(hl*8 + r)*32 + lm]      = (_Float16)p0[r];
            Pw[(hl*8 + r)*32 + 16 + lm] = (_Float16)p1[r];
        }
        v16h ap = cat8(*(const v8h*)(Pw + lm*32 + hl*16),
                       *(const v8h*)(Pw + lm*32 + hl*16 + 8));

        // O += P * V ; Lacc += P * ones (row sums on the matrix pipe)
        #pragma unroll
        for (int c = 0; c < 4; ++c) {
            v16h bv = *(const v16h*)(Vc + (c*16 + lm)*32 + hl*16);
            O[c] = wmma_f16(ap, bv, O[c]);
        }
        Lacc = wmma_f16(ap, ones, Lacc);
        __syncthreads();   // all waves done reading buf before it's refilled
    }

    // Normalize and scatter to [b, t, h*64+d] f16 for the output projection.
    float rinv[8];
    #pragma unroll
    for (int r = 0; r < 8; ++r) rinv[r] = 1.0f / Lacc[r];
    #pragma unroll
    for (int c = 0; c < 4; ++c)
        #pragma unroll
        for (int r = 0; r < 8; ++r) {
            const int qrow = qbase + hl*8 + r;
            attn16[((size_t)(bb*T + qrow))*D + hh*HD + c*16 + lm] =
                (_Float16)(O[c][r] * rinv[r]);
        }
}

// ---------------------------------------------------------------------------
// Workspace layout (bytes)
// ---------------------------------------------------------------------------
constexpr size_t SZ_X = (size_t)BT * D * 2;   // 8 MiB per activation tensor
constexpr size_t SZ_W = (size_t)D * D * 2;    // 2 MiB per weight
constexpr size_t OFF_XQ = 0;
constexpr size_t OFF_XK = OFF_XQ + SZ_X;
constexpr size_t OFF_XV = OFF_XK + SZ_X;
constexpr size_t OFF_WQ = OFF_XV + SZ_X;
constexpr size_t OFF_WK = OFF_WQ + SZ_W;
constexpr size_t OFF_WV = OFF_WK + SZ_W;
constexpr size_t OFF_WO = OFF_WV + SZ_W;
constexpr size_t OFF_Q  = OFF_WO + SZ_W;
constexpr size_t OFF_K  = OFF_Q  + SZ_X;
constexpr size_t OFF_V  = OFF_K  + SZ_X;
constexpr size_t OFF_AT = OFF_V  + SZ_X;
constexpr size_t OFF_SH = OFF_AT + SZ_X;
constexpr size_t OFF_IV = OFF_SH + (size_t)BT * 4;

extern "C" void kernel_launch(void* const* d_in, const int* in_sizes, int n_in,
                              void* d_out, int out_size, void* d_ws, size_t ws_size,
                              hipStream_t stream) {
    (void)in_sizes; (void)n_in; (void)out_size; (void)ws_size;
    const float* query  = (const float*)d_in[0];
    const float* key    = (const float*)d_in[1];
    const float* value  = (const float*)d_in[2];
    const float* phi    = (const float*)d_in[3];
    const float* logvar = (const float*)d_in[4];
    const float* Wq     = (const float*)d_in[5];
    const float* bq     = (const float*)d_in[6];
    const float* Wk     = (const float*)d_in[7];
    const float* bk     = (const float*)d_in[8];
    const float* Wv     = (const float*)d_in[9];
    const float* bv     = (const float*)d_in[10];
    const float* Wo     = (const float*)d_in[11];
    const float* bo     = (const float*)d_in[12];
    const float* wsig   = (const float*)d_in[13];
    const float* wphi   = (const float*)d_in[14];

    char* ws = (char*)d_ws;
    _Float16* xq   = (_Float16*)(ws + OFF_XQ);
    _Float16* xk   = (_Float16*)(ws + OFF_XK);
    _Float16* xv   = (_Float16*)(ws + OFF_XV);
    _Float16* wq16 = (_Float16*)(ws + OFF_WQ);
    _Float16* wk16 = (_Float16*)(ws + OFF_WK);
    _Float16* wv16 = (_Float16*)(ws + OFF_WV);
    _Float16* wo16 = (_Float16*)(ws + OFF_WO);
    _Float16* q16  = (_Float16*)(ws + OFF_Q);
    _Float16* k16  = (_Float16*)(ws + OFF_K);
    _Float16* vT16 = (_Float16*)(ws + OFF_V);
    _Float16* at16 = (_Float16*)(ws + OFF_AT);
    float*    shv  = (float*)(ws + OFF_SH);
    float*    ivv  = (float*)(ws + OFF_IV);

    // 1) Precision conversion (f32 -> f16)
    const int nAct4 = BT * D / 4;     // 1M float4 per activation tensor
    const int nW4   = D * D / 4;      // 256K float4 per weight
    cvt_f16x4<<<(nAct4 + 255)/256, 256, 0, stream>>>((const float4*)query, (v4h*)xq,   nAct4);
    cvt_f16x4<<<(nAct4 + 255)/256, 256, 0, stream>>>((const float4*)key,   (v4h*)xk,   nAct4);
    cvt_f16x4<<<(nAct4 + 255)/256, 256, 0, stream>>>((const float4*)value, (v4h*)xv,   nAct4);
    cvt_f16x4<<<(nW4   + 255)/256, 256, 0, stream>>>((const float4*)Wq,    (v4h*)wq16, nW4);
    cvt_f16x4<<<(nW4   + 255)/256, 256, 0, stream>>>((const float4*)Wk,    (v4h*)wk16, nW4);
    cvt_f16x4<<<(nW4   + 255)/256, 256, 0, stream>>>((const float4*)Wv,    (v4h*)wv16, nW4);
    cvt_f16x4<<<(nW4   + 255)/256, 256, 0, stream>>>((const float4*)Wo,    (v4h*)wo16, nW4);

    // 2) Per-token softmax bias scalars
    token_stats<<<BT/256, 256, 0, stream>>>(phi, logvar, wsig, wphi, shv, ivv);

    // 3) Q/K/V projections (WMMA GEMM). 2048 waves -> 512 blocks of 4 waves.
    gemm_kernel<0><<<512, 128, 0, stream>>>(xq, wq16, bq, q16,  nullptr);
    gemm_kernel<0><<<512, 128, 0, stream>>>(xk, wk16, bk, k16,  nullptr);
    gemm_kernel<1><<<512, 128, 0, stream>>>(xv, wv16, bv, vT16, nullptr);

    // 4) Flash attention with per-token shift/scale softmax.
    attn_kernel<<<Bsz*H*(T/64), 128, 0, stream>>>(q16, k16, vT16, shv, ivv, at16);

    // 5) Output projection -> f32 d_out
    gemm_kernel<2><<<512, 128, 0, stream>>>(at16, wo16, bo, nullptr, (float*)d_out);
}
